// GRUSeq2Seq_57801669870212
// MI455X (gfx1250) — matrix operand order repaired and checked by reading, not compiled, not run on hardware
//
#include <hip/hip_runtime.h>

// ---------------------------------------------------------------------------
// Fused persistent GRU seq2seq for MI455X (gfx1250), wave32 + WMMA bf16.
//   grid = 128 WGs x 256 threads (8 waves); each WG owns 16 batch rows.
//   LDS: bf16 weight cache (reloaded at enc->dec phase switch) + h state.
// ---------------------------------------------------------------------------

#define T_IN   336
#define NF     16
#define H      128
#define T_OUT  168
#define ROWS   16
#define BLOCK  256
#define NB     2048

typedef __attribute__((ext_vector_type(16))) __bf16 v16bf;
typedef __attribute__((ext_vector_type(8)))  float  v8f;

union Frag {
    v16bf bf;
    uint4 q[2];
    unsigned short h[16];
};

// ----- LDS layout (bytes); total 325,760 <= 320 KiB --------------------------
#define OFF_WA   0                   // 384x128 bf16: enc_Whh0 | dec_Whh0
#define OFF_WB   98304               // 384x128 bf16: enc_Wih1 | dec_Wih1
#define OFF_WC   196608              // 384x128 bf16: enc_Whh1 | dec_Whh1
#define OFF_WD   294912              // enc: Wih0 (12288B) | dec: fc_W1 (16384B)
#define OFF_HID  (OFF_WD + 16384)    // 16x64 bf16 hidden of FC1 (decoder)
#define OFF_PRED (OFF_HID + 2048)    // 16 f32 preds (decoder) / zero pad (encoder)
#define OFF_H0   (OFF_WD + 18560)    // double-buffered 2 x (16x128 bf16)
#define OFF_H1   (OFF_H0 + 8192)     // 16x128 bf16
#define LDS_BYTES (OFF_H1 + 4096)    // = 325760

__device__ __forceinline__ unsigned f2bf1(float f) {          // f32 -> bf16 (RNE)
    unsigned u = __builtin_bit_cast(unsigned, f);
    return (u + 0x7fffu + ((u >> 16) & 1u)) >> 16;
}
__device__ __forceinline__ unsigned short f2bfh(float f) { return (unsigned short)f2bf1(f); }
__device__ __forceinline__ float sigm(float x) { return 1.0f / (1.0f + __expf(-x)); }

__device__ __forceinline__ v8f wmma_bf16(const Frag& a, const Frag& b, v8f c) {
    return __builtin_amdgcn_wmma_f32_16x16x32_bf16(false, a.bf, false, b.bf,
                                                   (short)0, c, false, false);
}

// A operand (16 rows x K32 chunk) from row-major bf16, row stride R halves.
// Lane layout per ISA 7.12.2: lanes<16 hold K {k0..k0+7, k0+16..k0+23},
// lanes>=16 hold K {k0+8..k0+15, k0+24..k0+31}, row = lane%16.
__device__ __forceinline__ Frag loadA(const unsigned short* base, int m, int hi, int k0, int R) {
    Frag f;
    const unsigned short* p = base + m * R + k0 + hi * 8;
    f.q[0] = *(const uint4*)(p);
    f.q[1] = *(const uint4*)(p + 16);
    return f;
}
// B operand (K32 x 16) for C = h @ W^T: B[k][n] = W[n][k]. Lane holds column
// n = lane%16; lanes<16 carry K {k0..k0+7, k0+16..k0+23}, lanes>=16 carry K {k0+8..k0+15, k0+24..k0+31}.
__device__ __forceinline__ Frag loadB(const unsigned short* W, int col, int hi, int k0, int R) {
    Frag f;
    const unsigned short* p = W + col * R + k0 + hi * 8;
    f.q[0] = *(const uint4*)(p);
    f.q[1] = *(const uint4*)(p + 16);
    return f;
}

// cooperative f32 global -> bf16 LDS weight conversion
__device__ __forceinline__ void cvt_store(unsigned short* dst, const float* src, int n, int tid) {
    for (int i = tid * 2; i < n; i += BLOCK * 2) {
        float2 v = *(const float2*)(src + i);
        *(unsigned*)(dst + i) = f2bf1(v.x) | (f2bf1(v.y) << 16);
    }
}

__global__ __launch_bounds__(BLOCK, 1)
void gru_s2s_kernel(const float* __restrict__ x,
                    const float* __restrict__ eWih0, const float* __restrict__ eWhh0,
                    const float* __restrict__ ebih0, const float* __restrict__ ebhh0,
                    const float* __restrict__ eWih1, const float* __restrict__ eWhh1,
                    const float* __restrict__ ebih1, const float* __restrict__ ebhh1,
                    const float* __restrict__ dWih0, const float* __restrict__ dWhh0,
                    const float* __restrict__ dbih0, const float* __restrict__ dbhh0,
                    const float* __restrict__ dWih1, const float* __restrict__ dWhh1,
                    const float* __restrict__ dbih1, const float* __restrict__ dbhh1,
                    const float* __restrict__ fcW1, const float* __restrict__ fcb1,
                    const float* __restrict__ fcW2, const float* __restrict__ fcb2,
                    float* __restrict__ out)
{
    extern __shared__ char smem[];
    unsigned short* WA  = (unsigned short*)(smem + OFF_WA);
    unsigned short* WB  = (unsigned short*)(smem + OFF_WB);
    unsigned short* WC  = (unsigned short*)(smem + OFF_WC);
    unsigned short* WD  = (unsigned short*)(smem + OFF_WD);
    unsigned short* HID = (unsigned short*)(smem + OFF_HID);
    float*          predv  = (float*)(smem + OFF_PRED);
    unsigned short* zpad   = (unsigned short*)(smem + OFF_PRED); // 64B of zeros in enc
    unsigned short* H0base = (unsigned short*)(smem + OFF_H0);   // 2 buffers x 2048 halves
    unsigned short* H1b    = (unsigned short*)(smem + OFF_H1);

    const int tid  = threadIdx.x;
    const int wave = tid >> 5;
    const int lane = tid & 31;
    const int m    = lane & 15;     // row (A) or column-in-tile (B/C)
    const int hi   = lane >> 4;
    const int c    = wave * 16 + m; // this wave's hidden column (0..127)
    const int row0 = blockIdx.x * ROWS;

    // zero h state + pred/zero-pad area
    for (int i = tid; i < (8192 + 4096) / 4; i += BLOCK)
        ((unsigned*)(smem + OFF_H0))[i] = 0u;
    if (tid < 16) predv[tid] = 0.0f;

    // ---- load encoder weights into LDS as bf16 ----
    cvt_store(WA, eWhh0, 3 * H * H, tid);
    cvt_store(WB, eWih1, 3 * H * H, tid);
    cvt_store(WC, eWhh1, 3 * H * H, tid);
    cvt_store(WD, eWih0, 3 * H * NF, tid);
    __syncthreads();

    // hoisted per-lane biases (constant over time)
    float bi0[3], bh0[3], bi1[3], bh1[3];
#pragma unroll
    for (int g = 0; g < 3; ++g) {
        bi0[g] = ebih0[g * H + c]; bh0[g] = ebhh0[g * H + c];
        bi1[g] = ebih1[g * H + c]; bh1[g] = ebhh1[g * H + c];
    }

    const v8f vzero = {0.f, 0.f, 0.f, 0.f, 0.f, 0.f, 0.f, 0.f};
    v8f h0r = vzero, h1r = vzero;   // f32 hidden state, C-matrix lane layout

    // ======================= ENCODER: fused 2-layer GRU ======================
#pragma unroll 1
    for (int t = 0; t < T_IN; ++t) {
        __syncthreads();
        const unsigned short* H0p = H0base + (((t + 1) & 1) << 11);  // prev h0
        unsigned short*       H0n = H0base + ((t & 1) << 11);        // new  h0

        // ---- layer 0: gi = x_t @ Wih0^T (K=16 padded to 32) ----
        Frag ax;
        {
            const float* xp = x + ((size_t)(row0 + m) * T_IN + t) * NF + hi * 8;
            float4 a = *(const float4*)xp;
            float4 b = *(const float4*)(xp + 4);
            ax.q[0] = make_uint4(f2bf1(a.x) | (f2bf1(a.y) << 16),
                                 f2bf1(a.z) | (f2bf1(a.w) << 16),
                                 f2bf1(b.x) | (f2bf1(b.y) << 16),
                                 f2bf1(b.z) | (f2bf1(b.w) << 16));
            ax.q[1] = make_uint4(0, 0, 0, 0);
            if (t + 1 < T_IN) __builtin_prefetch(xp + NF, 0, 1);  // global_prefetch_b8
        }
        v8f gi[3], gh[3];
#pragma unroll
        for (int g = 0; g < 3; ++g) {
            // lo lanes read Wih0 row K 0..7, hi lanes K 8..15; elements 8..15
            // (K 16..31 of the padded tile) are zero.
            const unsigned short* p = WD + (g * H + c) * NF + hi * 8;
            Frag b;
            b.q[0] = *(const uint4*)(p);
            b.q[1] = make_uint4(0, 0, 0, 0);
            gi[g] = wmma_bf16(ax, b, vzero);
            gh[g] = vzero;
        }
        // gh = h0 @ Whh0^T
#pragma unroll
        for (int kk = 0; kk < 4; ++kk) {
            Frag a = loadA(H0p, m, hi, kk * 32, H);
#pragma unroll
            for (int g = 0; g < 3; ++g) {
                Frag b = loadB(WA, g * H + c, hi, kk * 32, H);
                gh[g] = wmma_bf16(a, b, gh[g]);
            }
        }
#pragma unroll
        for (int e = 0; e < 8; ++e) {
            float r = sigm(gi[0][e] + bi0[0] + gh[0][e] + bh0[0]);
            float z = sigm(gi[1][e] + bi0[1] + gh[1][e] + bh0[1]);
            float n = tanhf(gi[2][e] + bi0[2] + r * (gh[2][e] + bh0[2]));
            h0r[e] = (1.f - z) * n + z * h0r[e];
        }
#pragma unroll
        for (int e = 0; e < 8; ++e)
            H0n[(e + hi * 8) * H + c] = f2bfh(h0r[e]);
        __syncthreads();

        // ---- layer 1: gi1 = h0_new @ Wih1^T ; gh1 = h1 @ Whh1^T ----
        v8f gi1v[3] = {vzero, vzero, vzero}, gh1v[3] = {vzero, vzero, vzero};
#pragma unroll
        for (int kk = 0; kk < 4; ++kk) {
            Frag a0 = loadA(H0n, m, hi, kk * 32, H);
            Frag a1 = loadA(H1b, m, hi, kk * 32, H);
#pragma unroll
            for (int g = 0; g < 3; ++g) {
                Frag wi = loadB(WB, g * H + c, hi, kk * 32, H);
                gi1v[g] = wmma_bf16(a0, wi, gi1v[g]);
                Frag wh = loadB(WC, g * H + c, hi, kk * 32, H);
                gh1v[g] = wmma_bf16(a1, wh, gh1v[g]);
            }
        }
#pragma unroll
        for (int e = 0; e < 8; ++e) {
            float r = sigm(gi1v[0][e] + bi1[0] + gh1v[0][e] + bh1[0]);
            float z = sigm(gi1v[1][e] + bi1[1] + gh1v[1][e] + bh1[1]);
            float n = tanhf(gi1v[2][e] + bi1[2] + r * (gh1v[2][e] + bh1[2]));
            h1r[e] = (1.f - z) * n + z * h1r[e];
        }
        __syncthreads();                         // WAR: H1 readers done
#pragma unroll
        for (int e = 0; e < 8; ++e)
            H1b[(e + hi * 8) * H + c] = f2bfh(h1r[e]);
    }

    // ================== phase switch: load decoder weights ==================
    __syncthreads();
    cvt_store(WA, dWhh0, 3 * H * H, tid);
    cvt_store(WB, dWih1, 3 * H * H, tid);
    cvt_store(WC, dWhh1, 3 * H * H, tid);
    cvt_store(WD, fcW1, (H / 2) * H, tid);
    __syncthreads();

    float dbi0[3], dbh0[3], dbi1[3], dbh1[3], w0[3];
#pragma unroll
    for (int g = 0; g < 3; ++g) {
        dbi0[g] = dbih0[g * H + c]; dbh0[g] = dbhh0[g * H + c];
        dbi1[g] = dbih1[g * H + c]; dbh1[g] = dbhh1[g * H + c];
        w0[g]   = dWih0[g * H + c];             // input dim is 1 -> outer product
    }
    const float fb1 = (wave < 4) ? fcb1[wave * 16 + m] : 0.f;
    const float fb2 = fcb2[0];
    Frag bw2[2];                                // fc_W2 as B fragments (only col 0)
#pragma unroll
    for (int kk = 0; kk < 2; ++kk)
#pragma unroll
        for (int i = 0; i < 16; ++i)
            bw2[kk].h[i] = f2bfh((m == 0) ? fcW2[kk * 32 + (i >> 3) * 16 + hi * 8 + (i & 7)] : 0.f);

    // ================= DECODER: autoregressive 2-layer GRU + FC =============
#pragma unroll 1
    for (int s = 0; s < T_OUT; ++s) {
        __syncthreads();
        const unsigned short* H0p = H0base + (((s + 1) & 1) << 11); // enc ended on buf 1
        unsigned short*       H0n = H0base + ((s & 1) << 11);

        float pr[8];
        {
            float4 p0 = *(const float4*)(predv + hi * 8);
            float4 p1 = *(const float4*)(predv + hi * 8 + 4);
            pr[0] = p0.x; pr[1] = p0.y; pr[2] = p0.z; pr[3] = p0.w;
            pr[4] = p1.x; pr[5] = p1.y; pr[6] = p1.z; pr[7] = p1.w;
        }
        // layer 0: gh via WMMA, gi via scalar outer product pred*Wih0
        v8f gh[3] = {vzero, vzero, vzero};
#pragma unroll
        for (int kk = 0; kk < 4; ++kk) {
            Frag a = loadA(H0p, m, hi, kk * 32, H);
#pragma unroll
            for (int g = 0; g < 3; ++g) {
                Frag b = loadB(WA, g * H + c, hi, kk * 32, H);
                gh[g] = wmma_bf16(a, b, gh[g]);
            }
        }
#pragma unroll
        for (int e = 0; e < 8; ++e) {
            float r = sigm(pr[e] * w0[0] + dbi0[0] + gh[0][e] + dbh0[0]);
            float z = sigm(pr[e] * w0[1] + dbi0[1] + gh[1][e] + dbh0[1]);
            float n = tanhf(pr[e] * w0[2] + dbi0[2] + r * (gh[2][e] + dbh0[2]));
            h0r[e] = (1.f - z) * n + z * h0r[e];
        }
#pragma unroll
        for (int e = 0; e < 8; ++e)
            H0n[(e + hi * 8) * H + c] = f2bfh(h0r[e]);
        __syncthreads();

        // layer 1
        v8f gi1v[3] = {vzero, vzero, vzero}, gh1v[3] = {vzero, vzero, vzero};
#pragma unroll
        for (int kk = 0; kk < 4; ++kk) {
            Frag a0 = loadA(H0n, m, hi, kk * 32, H);
            Frag a1 = loadA(H1b, m, hi, kk * 32, H);
#pragma unroll
            for (int g = 0; g < 3; ++g) {
                Frag wi = loadB(WB, g * H + c, hi, kk * 32, H);
                gi1v[g] = wmma_bf16(a0, wi, gi1v[g]);
                Frag wh = loadB(WC, g * H + c, hi, kk * 32, H);
                gh1v[g] = wmma_bf16(a1, wh, gh1v[g]);
            }
        }
#pragma unroll
        for (int e = 0; e < 8; ++e) {
            float r = sigm(gi1v[0][e] + dbi1[0] + gh1v[0][e] + dbh1[0]);
            float z = sigm(gi1v[1][e] + dbi1[1] + gh1v[1][e] + dbh1[1]);
            float n = tanhf(gi1v[2][e] + dbi1[2] + r * (gh1v[2][e] + dbh1[2]));
            h1r[e] = (1.f - z) * n + z * h1r[e];
        }
        __syncthreads();                         // WAR on H1
#pragma unroll
        for (int e = 0; e < 8; ++e)
            H1b[(e + hi * 8) * H + c] = f2bfh(h1r[e]);
        __syncthreads();                         // H1 new visible for FC1

        // FC1: hid = relu(h1 @ fcW1^T + b1), N=64 -> waves 0..3, one tile each
        if (wave < 4) {
            v8f hacc = vzero;
#pragma unroll
            for (int kk = 0; kk < 4; ++kk) {
                Frag a = loadA(H1b, m, hi, kk * 32, H);
                Frag b = loadB(WD, wave * 16 + m, hi, kk * 32, H);
                hacc = wmma_bf16(a, b, hacc);
            }
#pragma unroll
            for (int e = 0; e < 8; ++e)
                HID[(e + hi * 8) * 64 + wave * 16 + m] = f2bfh(fmaxf(hacc[e] + fb1, 0.f));
        }
        __syncthreads();

        // FC2: pred = relu(hid @ fcW2^T + b2) -> wave 0, result in column 0
        if (wave == 0) {
            v8f acc = vzero;
#pragma unroll
            for (int kk = 0; kk < 2; ++kk) {
                Frag a = loadA(HID, m, hi, kk * 32, 64);
                acc = wmma_bf16(a, bw2[kk], acc);
            }
            if (m == 0) {
#pragma unroll
                for (int e = 0; e < 8; ++e) {
                    float p = fmaxf(acc[e] + fb2, 0.f);
                    int row = e + hi * 8;
                    predv[row] = p;
                    out[(size_t)(row0 + row) * T_OUT + s] = p;
                }
            }
        }
    }
}

extern "C" void kernel_launch(void* const* d_in, const int* in_sizes, int n_in,
                              void* d_out, int out_size, void* d_ws, size_t ws_size,
                              hipStream_t stream) {
    (void)in_sizes; (void)n_in; (void)out_size; (void)d_ws; (void)ws_size;
    const float* x     = (const float*)d_in[0];
    const float* eWih0 = (const float*)d_in[1];
    const float* eWhh0 = (const float*)d_in[2];
    const float* ebih0 = (const float*)d_in[3];
    const float* ebhh0 = (const float*)d_in[4];
    const float* eWih1 = (const float*)d_in[5];
    const float* eWhh1 = (const float*)d_in[6];
    const float* ebih1 = (const float*)d_in[7];
    const float* ebhh1 = (const float*)d_in[8];
    const float* dWih0 = (const float*)d_in[9];
    const float* dWhh0 = (const float*)d_in[10];
    const float* dbih0 = (const float*)d_in[11];
    const float* dbhh0 = (const float*)d_in[12];
    const float* dWih1 = (const float*)d_in[13];
    const float* dWhh1 = (const float*)d_in[14];
    const float* dbih1 = (const float*)d_in[15];
    const float* dbhh1 = (const float*)d_in[16];
    const float* fcW1  = (const float*)d_in[17];
    const float* fcb1  = (const float*)d_in[18];
    const float* fcW2  = (const float*)d_in[19];
    const float* fcb2  = (const float*)d_in[20];
    float* out = (float*)d_out;

    // opt in to >64KB dynamic LDS (idempotent, not a stream op -> capture-safe)
    (void)hipFuncSetAttribute((const void*)gru_s2s_kernel,
                              hipFuncAttributeMaxDynamicSharedMemorySize, LDS_BYTES);

    gru_s2s_kernel<<<dim3(NB / ROWS), dim3(BLOCK), LDS_BYTES, stream>>>(
        x, eWih0, eWhh0, ebih0, ebhh0, eWih1, eWhh1, ebih1, ebhh1,
        dWih0, dWhh0, dbih0, dbhh0, dWih1, dWhh1, dbih1, dbhh1,
        fcW1, fcb1, fcW2, fcb2, out);
}
